// GATClassifier_89773406421652
// MI455X (gfx1250) — matrix operand order repaired and checked by reading, not compiled, not run on hardware
//
#include <hip/hip_runtime.h>
#include <hip/hip_bf16.h>

typedef __attribute__((ext_vector_type(16))) __bf16 v16bf;
typedef __attribute__((ext_vector_type(8)))  float  v8f;

// ---------------------------------------------------------------------------
// helpers
// ---------------------------------------------------------------------------
__device__ __forceinline__ unsigned enc_f32(float x) {
    unsigned u = __float_as_uint(x);
    return (u & 0x80000000u) ? ~u : (u | 0x80000000u);   // monotone encoding
}
__device__ __forceinline__ float dec_f32(unsigned u) {
    unsigned v = (u & 0x80000000u) ? (u & 0x7FFFFFFFu) : ~u;
    return __uint_as_float(v);
}

// ---------------------------------------------------------------------------
// zero-fill (graph-capture-safe replacement for hipMemsetAsync)
// ---------------------------------------------------------------------------
__global__ void gat_zero(float* __restrict__ p, size_t n) {
    size_t t = (size_t)blockIdx.x * blockDim.x + threadIdx.x;
    if (t < n) p[t] = 0.0f;
}

// ---------------------------------------------------------------------------
// Pack f32 weight W[K,M] into bf16 WMMA B-fragment order.
// Fragment for (colTile ct, kTile kt): 32 lanes x 16 bf16, lane-contiguous:
//   Wp[(((ct*kTiles)+kt)*32 + lane)*16 + {2v,2v+1}] =
//       W[(kt*32 + (lane>>4)*8 + (v>=4?16:0) + 2(v&3))*M + ct*16 + (lane&15)]
// so the GEMM loads each lane's fragment as 32 contiguous bytes (2x b128).
// ---------------------------------------------------------------------------
__global__ void gat_packW(const float* __restrict__ W, __bf16* __restrict__ Wp,
                          int K, int M)
{
    int t = blockIdx.x * blockDim.x + threadIdx.x;
    int kTiles = K >> 5;
    int cTiles = M >> 4;
    int tot = cTiles * kTiles * 32;
    if (t >= tot) return;
    int lane = t & 31;
    int kt   = (t >> 5) % kTiles;
    int ct   = (t >> 5) / kTiles;
    int half = lane >> 4;
    int c    = lane & 15;
    __bf16* dst = Wp + (size_t)t * 16;
    const float* srcp = W + (size_t)(kt * 32 + half * 8) * M + ct * 16 + c;
    #pragma unroll
    for (int v = 0; v < 8; ++v) {
        dst[2 * v]     = (__bf16)srcp[(size_t)(((v & 4) ? 16 : 0) + ((v & 3) << 1)) * M];
        dst[2 * v + 1] = (__bf16)srcp[(size_t)(((v & 4) ? 16 : 0) + ((v & 3) << 1) + 1) * M];
    }
}

// ---------------------------------------------------------------------------
// WMMA GEMM:  Y[Nrows,M] = X[Nrows,K] (f32 -> bf16 on the fly) @ W (packed bf16)
// One wave per block; wave computes a 16x64 output tile with 4 accumulators so
// the A fragment (loads + cvt) is amortized over 4 v_wmma ops.
// Requires Nrows%16==0, K%32==0, M%64==0 (grid exact; EXEC all ones).
// ---------------------------------------------------------------------------
__global__ __launch_bounds__(32) void gat_gemm_wmma(
        const float* __restrict__ X, const __bf16* __restrict__ Wp,
        float* __restrict__ Y, int Nrows, int K, int M)
{
    const int lane = threadIdx.x & 31;
    const int r    = lane & 15;
    const int half = lane >> 4;                 // 0: lanes 0-15, 1: lanes 16-31

    const int rowBase = blockIdx.x * 16;
    const int colBase = blockIdx.y * 64;
    const int ct0     = colBase >> 4;           // first of 4 col tiles
    const int kTiles  = K >> 5;

    const int arow = rowBase + r;               // A row this lane feeds

    v8f acc[4] = {};
    for (int kt = 0; kt < kTiles; ++kt) {
        const int kb = kt << 5;
        // ---- A fragment: 16x32 bf16, ISA layout:
        // lanes 0-15 : VGPR v<4 -> K=2v,2v+1 ; v>=4 -> K=16+2(v-4),..
        // lanes 16-31: same +8
        v16bf a;
        const float* xrow = X + (size_t)arow * K + kb + half * 8;
        #pragma unroll
        for (int v = 0; v < 8; ++v) {
            int k = ((v & 4) ? 16 : 0) + ((v & 3) << 1);
            a[2 * v]     = (__bf16)xrow[k];
            a[2 * v + 1] = (__bf16)xrow[k + 1];
        }
        // ---- 4 packed B fragments, each a 32B contiguous per-lane load
        #pragma unroll
        for (int i = 0; i < 4; ++i) {
            const v16bf b = *(const v16bf*)(
                Wp + ((size_t)((ct0 + i) * kTiles + kt) * 32 + lane) * 16);
            acc[i] = __builtin_amdgcn_wmma_f32_16x16x32_bf16(
                         false, a, false, b, (short)0, acc[i], false, false);
        }
    }
    // ---- store C: lane col = r, VGPR v -> row = v + half*8
    #pragma unroll
    for (int i = 0; i < 4; ++i) {
        float* yb = Y + (size_t)(rowBase + half * 8) * M + colBase + i * 16 + r;
        #pragma unroll
        for (int v = 0; v < 8; ++v)
            yb[(size_t)v * M] = acc[i][v];
    }
}

// ---------------------------------------------------------------------------
// per-(node,head) attention logits: as_n = <h[n,h,:], a_src[h,:]>, same for dst
// one wave per (node,head)
// ---------------------------------------------------------------------------
__global__ void gat_scores(const float* __restrict__ Hm,
                           const float* __restrict__ a_src,
                           const float* __restrict__ a_dst,
                           float* __restrict__ asn, float* __restrict__ adn,
                           int n, int Hh, int C)
{
    int gwave = (int)(((size_t)blockIdx.x * blockDim.x + threadIdx.x) >> 5);
    int lane  = threadIdx.x & 31;
    int node  = gwave / Hh;
    int h     = gwave - node * Hh;
    if (node >= n) return;
    int F = Hh * C;
    float s0 = 0.f, s1 = 0.f;
    for (int c = lane; c < C; c += 32) {
        float hv = Hm[(size_t)node * F + h * C + c];
        s0 += hv * a_src[h * C + c];
        s1 += hv * a_dst[h * C + c];
    }
    #pragma unroll
    for (int o = 16; o > 0; o >>= 1) {
        s0 += __shfl_xor(s0, o, 32);
        s1 += __shfl_xor(s1, o, 32);
    }
    if (lane == 0) {
        asn[node * Hh + h] = s0;
        adn[node * Hh + h] = s1;
    }
}

// ---------------------------------------------------------------------------
// edge pass 1: segment max via atomic umax on order-preserving encoding
// thread per (edge_ext, head); edges E.. E+N-1 are self loops
// ---------------------------------------------------------------------------
__global__ void gat_edge_max(const int* __restrict__ src, const int* __restrict__ dst,
                             const float* __restrict__ asn, const float* __restrict__ adn,
                             unsigned* __restrict__ mx, int E_, int n, int Hh)
{
    int t = blockIdx.x * blockDim.x + threadIdx.x;
    int tot = (E_ + n) * Hh;
    if (t >= tot) return;
    int e = t / Hh, h = t - e * Hh;
    int s = (e < E_) ? src[e] : (e - E_);
    int d = (e < E_) ? dst[e] : (e - E_);
    float v = asn[s * Hh + h] + adn[d * Hh + h];
    v = (v > 0.f) ? v : v * 0.2f;                 // leaky relu 0.2
    atomicMax(&mx[d * Hh + h], enc_f32(v));
}

// ---------------------------------------------------------------------------
// edge pass 2: ex = exp(e - m[dst]); denom[dst] += ex
// ---------------------------------------------------------------------------
__global__ void gat_edge_exp(const int* __restrict__ src, const int* __restrict__ dst,
                             const float* __restrict__ asn, const float* __restrict__ adn,
                             const unsigned* __restrict__ mx,
                             float* __restrict__ exbuf, float* __restrict__ denom,
                             int E_, int n, int Hh)
{
    int t = blockIdx.x * blockDim.x + threadIdx.x;
    int tot = (E_ + n) * Hh;
    if (t >= tot) return;
    int e = t / Hh, h = t - e * Hh;
    int s = (e < E_) ? src[e] : (e - E_);
    int d = (e < E_) ? dst[e] : (e - E_);
    float v = asn[s * Hh + h] + adn[d * Hh + h];
    v = (v > 0.f) ? v : v * 0.2f;
    float m = dec_f32(mx[d * Hh + h]);
    float ex = __expf(v - m);
    exbuf[t] = ex;
    atomicAdd(&denom[d * Hh + h], ex);
}

// ---------------------------------------------------------------------------
// edge pass 3: out[dst] += alpha * h[src]   (wave per edge, 32-lane strips)
// ---------------------------------------------------------------------------
__global__ void gat_edge_agg(const int* __restrict__ src, const int* __restrict__ dst,
                             const float* __restrict__ exbuf, const float* __restrict__ denom,
                             const float* __restrict__ Hm, float* __restrict__ out,
                             int E_, int n, int Hh, int C)
{
    int gwave = (int)(((size_t)blockIdx.x * blockDim.x + threadIdx.x) >> 5);
    int lane  = threadIdx.x & 31;
    int tot   = E_ + n;
    if (gwave >= tot) return;
    int e = gwave;
    int s = (e < E_) ? src[e] : (e - E_);
    int d = (e < E_) ? dst[e] : (e - E_);
    int F = Hh * C;
    for (int j = lane; j < F; j += 32) {
        int h = j / C;
        float alpha = exbuf[(size_t)e * Hh + h] / (denom[d * Hh + h] + 1e-16f);
        float v = Hm[(size_t)s * F + j];
        atomicAdd(&out[(size_t)d * F + j], alpha * v);
    }
}

// ---------------------------------------------------------------------------
// epilogue: y = elu(y + bias)
// ---------------------------------------------------------------------------
__global__ void gat_bias_elu(float* __restrict__ Y, const float* __restrict__ bias,
                             size_t total, int F)
{
    size_t t = (size_t)blockIdx.x * blockDim.x + threadIdx.x;
    if (t >= total) return;
    float v = Y[t] + bias[(int)(t % (size_t)F)];
    Y[t] = (v > 0.f) ? v : (__expf(v) - 1.f);
}

// ---------------------------------------------------------------------------
// global mean pool (sum + counts via atomics)
// ---------------------------------------------------------------------------
__global__ void gat_pool(const float* __restrict__ H3, const int* __restrict__ batch,
                         float* __restrict__ gsum, float* __restrict__ gcnt, int n, int C)
{
    int t = blockIdx.x * blockDim.x + threadIdx.x;
    if (t >= n * C) return;
    int node = t / C, c = t - node * C;
    int g = batch[node];
    atomicAdd(&gsum[g * C + c], H3[t]);
    if (c == 0) atomicAdd(&gcnt[g], 1.0f);
}

// ---------------------------------------------------------------------------
// MLP head
// ---------------------------------------------------------------------------
__global__ void gat_head1(const float* __restrict__ gsum, const float* __restrict__ gcnt,
                          const float* __restrict__ Wh1, const float* __restrict__ bh1,
                          float* __restrict__ z, int Gn, int C)
{
    int t = blockIdx.x * blockDim.x + threadIdx.x;
    if (t >= Gn * C) return;
    int g = t / C, j = t - g * C;
    float inv = 1.0f / fmaxf(gcnt[g], 1.0f);
    float acc = bh1[j];
    for (int k = 0; k < C; ++k)
        acc += (gsum[g * C + k] * inv) * Wh1[k * C + j];
    z[t] = fmaxf(acc, 0.0f);
}

__global__ void gat_head2(const float* __restrict__ z, const float* __restrict__ Wh2,
                          const float* __restrict__ bh2, float* __restrict__ out,
                          int Gn, int C, int NC)
{
    int t = blockIdx.x * blockDim.x + threadIdx.x;
    if (t >= Gn * NC) return;
    int g = t / NC, j = t - g * NC;
    float acc = bh2[j];
    for (int k = 0; k < C; ++k)
        acc += z[g * C + k] * Wh2[k * NC + j];
    out[t] = acc;
}

// ---------------------------------------------------------------------------
// host orchestration
// ---------------------------------------------------------------------------
extern "C" void kernel_launch(void* const* d_in, const int* in_sizes, int n_in,
                              void* d_out, int out_size, void* d_ws, size_t ws_size,
                              hipStream_t stream)
{
    const int IN_CH = 128, HID = 64, HEADS = 4, NUM_CLASSES = 10;
    const int Nn = in_sizes[0] / IN_CH;        // 50000
    const int Ee = in_sizes[1];                // 800000
    const int Gg = out_size / NUM_CLASSES;     // 64
    const int F2 = HEADS * HID;                // 256

    const float* x    = (const float*)d_in[0];
    const int*   src  = (const int*)d_in[1];
    const int*   dst  = (const int*)d_in[2];
    const int*   batch= (const int*)d_in[3];
    const float* W1   = (const float*)d_in[4];
    const float* as1  = (const float*)d_in[5];
    const float* ad1  = (const float*)d_in[6];
    const float* b1   = (const float*)d_in[7];
    const float* W2   = (const float*)d_in[8];
    const float* as2  = (const float*)d_in[9];
    const float* ad2  = (const float*)d_in[10];
    const float* b2   = (const float*)d_in[11];
    const float* Wp   = (const float*)d_in[12];
    const float* asp  = (const float*)d_in[13];
    const float* adp  = (const float*)d_in[14];
    const float* bp   = (const float*)d_in[15];
    const float* Wh1  = (const float*)d_in[16];
    const float* bh1  = (const float*)d_in[17];
    const float* Wh2  = (const float*)d_in[18];
    const float* bh2  = (const float*)d_in[19];
    float* out = (float*)d_out;

    // ---- workspace carve-out (256B aligned slabs) ----
    char* wp_ = (char*)d_ws;
    auto carve = [&](size_t bytes) -> void* {
        void* p = (void*)wp_;
        wp_ += (bytes + 255) & ~(size_t)255;
        return p;
    };
    float*    bufA  = (float*)   carve((size_t)Nn * F2 * 4);         // gemm outputs
    float*    bufB  = (float*)   carve((size_t)Nn * F2 * 4);         // aggregation outputs
    float*    asn   = (float*)   carve((size_t)Nn * HEADS * 4);
    float*    adn   = (float*)   carve((size_t)Nn * HEADS * 4);
    unsigned* mx    = (unsigned*)carve((size_t)Nn * HEADS * 4);
    float*    denom = (float*)   carve((size_t)Nn * HEADS * 4);
    float*    exbuf = (float*)   carve((size_t)(Ee + Nn) * HEADS * 4);
    float*    gsum  = (float*)   carve((size_t)Gg * HID * 4);
    float*    gcnt  = (float*)   carve((size_t)Gg * 4);
    float*    zbuf  = (float*)   carve((size_t)Gg * HID * 4);
    __bf16*   w1b   = (__bf16*)  carve((size_t)IN_CH * F2 * 2);
    __bf16*   w2b   = (__bf16*)  carve((size_t)F2 * F2 * 2);
    __bf16*   wpb   = (__bf16*)  carve((size_t)F2 * HID * 2);
    (void)ws_size; (void)n_in;

    // ---- weight conversion + fragment packing (recomputed every call) ----
    {
        int t1 = (F2 / 16) * (IN_CH / 32) * 32;
        gat_packW<<<(t1 + 255) / 256, 256, 0, stream>>>(W1, w1b, IN_CH, F2);
        int t2 = (F2 / 16) * (F2 / 32) * 32;
        gat_packW<<<(t2 + 255) / 256, 256, 0, stream>>>(W2, w2b, F2, F2);
        int t3 = (HID / 16) * (F2 / 32) * 32;
        gat_packW<<<(t3 + 255) / 256, 256, 0, stream>>>(Wp, wpb, F2, HID);
    }

    // ---- one GAT layer ----
    auto conv = [&](const float* Xin, int K, const __bf16* Wb, int F, int Hh, int C,
                    const float* a_s, const float* a_d, const float* bias) {
        dim3 gb(Nn / 16, F / 64);
        gat_gemm_wmma<<<gb, 32, 0, stream>>>(Xin, Wb, bufA, Nn, K, F);

        int waves = Nn * Hh;
        gat_scores<<<(waves * 32 + 255) / 256, 256, 0, stream>>>(
            bufA, a_s, a_d, asn, adn, Nn, Hh, C);

        size_t nh = (size_t)Nn * Hh;
        gat_zero<<<(unsigned)((nh + 255) / 256), 256, 0, stream>>>((float*)mx, nh);
        gat_zero<<<(unsigned)((nh + 255) / 256), 256, 0, stream>>>(denom, nh);
        size_t nf = (size_t)Nn * F;
        gat_zero<<<(unsigned)((nf + 255) / 256), 256, 0, stream>>>(bufB, nf);

        int tot = (Ee + Nn) * Hh;
        gat_edge_max<<<(tot + 255) / 256, 256, 0, stream>>>(
            src, dst, asn, adn, mx, Ee, Nn, Hh);
        gat_edge_exp<<<(tot + 255) / 256, 256, 0, stream>>>(
            src, dst, asn, adn, mx, exbuf, denom, Ee, Nn, Hh);

        size_t aggThreads = (size_t)(Ee + Nn) * 32;
        gat_edge_agg<<<(unsigned)((aggThreads + 255) / 256), 256, 0, stream>>>(
            src, dst, exbuf, denom, bufA, bufB, Ee, Nn, Hh, C);

        gat_bias_elu<<<(unsigned)((nf + 255) / 256), 256, 0, stream>>>(bufB, bias, nf, F);
    };

    // conv1: x[N,128] -> bufB[N,256]
    conv(x, IN_CH, w1b, F2, HEADS, HID, as1, ad1, b1);
    // conv2: bufB[N,256] -> bufB[N,256]   (gemm reads bufB before bufB is re-zeroed)
    conv(bufB, F2, w2b, F2, HEADS, HID, as2, ad2, b2);
    // proj: bufB[N,256] -> bufB[N,64], heads=1 (mean over 1 head == identity)
    conv(bufB, F2, wpb, HID, 1, HID, asp, adp, bp);

    // ---- global mean pool ----
    gat_zero<<<((Gg * HID) + 255) / 256, 256, 0, stream>>>(gsum, (size_t)Gg * HID);
    gat_zero<<<(Gg + 255) / 256, 256, 0, stream>>>(gcnt, (size_t)Gg);
    gat_pool<<<((Nn * HID) + 255) / 256, 256, 0, stream>>>(bufB, batch, gsum, gcnt, Nn, HID);

    // ---- MLP head ----
    gat_head1<<<((Gg * HID) + 255) / 256, 256, 0, stream>>>(gsum, gcnt, Wh1, bh1, zbuf, Gg, HID);
    gat_head2<<<((Gg * NUM_CLASSES) + 255) / 256, 256, 0, stream>>>(zbuf, Wh2, bh2, out, Gg, HID, NUM_CLASSES);
}
